// CrossAttentionBlock_38774964748965
// MI455X (gfx1250) — matrix operand order repaired and checked by reading, not compiled, not run on hardware
//
#include <hip/hip_runtime.h>
#include <hip/hip_bf16.h>

// ---------------- problem constants ----------------
constexpr int B_  = 16;
constexpr int C_  = 512;
constexpr int HW_ = 4096;   // 64*64
constexpr int P_  = 512;
constexpr int N_  = 128;
constexpr int D_  = 1024;
constexpr int ROWS = 32;    // query rows processed per workgroup in fused kernels
constexpr int SCH  = 8;     // softmax reduction chunks over the HW axis

typedef __attribute__((ext_vector_type(16))) __bf16 v16bf;
typedef __attribute__((ext_vector_type(8)))  float  v8f;

union BFrag {
    v16bf v;
    uint4 u[2];
    __bf16 e[16];
};

// fp32 -> bf16 native conversion (lowers to v_cvt_pk_bf16_f32)
__device__ __forceinline__ __bf16 f2bf(float f) { return (__bf16)f; }

__device__ __forceinline__ v8f wmma_bf16(const BFrag& a, const BFrag& b, v8f c) {
    return __builtin_amdgcn_wmma_f32_16x16x32_bf16(
        /*neg_a=*/false, a.v, /*neg_b=*/false, b.v,
        /*c_mod=*/(short)0, c, /*reuse_a=*/false, /*reuse_b=*/false);
}

// =====================================================================
// Kernel 0: tiled transpose + convert: (K x Nc fp32 row-major) -> bf16
// n-major out[n*K + k]. LDS 32x33 tile, coalesced both directions.
// grid(K/32, Nc/32), block(256)
// =====================================================================
__global__ void __launch_bounds__(256)
transpose_cvt(const float* __restrict__ in, __bf16* __restrict__ outp,
              int K, int Nc)
{
    __shared__ float t[32][33];
    const int k0 = blockIdx.x * 32, n0 = blockIdx.y * 32;
    const int tx = threadIdx.x & 31, ty = threadIdx.x >> 5;   // 8 rows/pass
#pragma unroll
    for (int r = 0; r < 32; r += 8)
        t[ty + r][tx] = in[(size_t)(k0 + ty + r) * Nc + n0 + tx];
    __syncthreads();
#pragma unroll
    for (int r = 0; r < 32; r += 8)
        outp[(size_t)(n0 + ty + r) * K + k0 + tx] = f2bf(t[tx][ty + r]);
}

// =====================================================================
// Kernel 1: k = y @ Wk  (bf16, row-major N x P  -> usable as B^T operand)
//           v = y @ Wv  (bf16, stored TRANSPOSED P x N for contiguous B frags)
// Weights pre-transposed bf16: wkt/wvt are (P x D), n-major.
// grid(32 = P/16, 8 = N/16, 32 = B*2), block(32) : one wave per 16x16 tile
// =====================================================================
__global__ void __launch_bounds__(32)
kv_kernel(const float* __restrict__ y, const __bf16* __restrict__ wkt,
          const __bf16* __restrict__ wvt, __bf16* __restrict__ kbf,
          __bf16* __restrict__ vt)
{
    const int n0  = blockIdx.x * 16;          // P column
    const int m0  = blockIdx.y * 16;          // N row
    const int b   = blockIdx.z >> 1;
    const bool isV = blockIdx.z & 1;
    const float*  __restrict__ A  = y + (size_t)b * N_ * D_;   // N x D, lda = D
    const __bf16* __restrict__ Bt = isV ? wvt : wkt;           // (P x D) n-major bf16

    const int lane = threadIdx.x & 31;
    const int mr = lane & 15, h = lane >> 4;

    v8f acc = {};
    for (int k0 = 0; k0 < D_; k0 += 32) {
        BFrag a, bb;
#pragma unroll
        for (int j = 0; j < 8; ++j) {
            a.e[j]     = f2bf(A[(size_t)(m0 + mr) * D_ + k0 + h * 8 + j]);
            a.e[8 + j] = f2bf(A[(size_t)(m0 + mr) * D_ + k0 + 16 + h * 8 + j]);
        }
        const uint4* bp = (const uint4*)(Bt + (size_t)(n0 + mr) * D_ + k0 + h * 8);
        bb.u[0] = bp[0];
        bb.u[1] = bp[2];
        acc = wmma_bf16(a, bb, acc);
    }
    if (!isV) {
#pragma unroll
        for (int r = 0; r < 8; ++r)
            kbf[(size_t)b * N_ * P_ + (size_t)(m0 + h * 8 + r) * P_ + n0 + mr] = f2bf(acc[r]);
    } else {
        // v[nrow][p] stored as vt[p][nrow]
#pragma unroll
        for (int r = 0; r < 8; ++r)
            vt[(size_t)b * P_ * N_ + (size_t)(n0 + mr) * N_ + m0 + h * 8 + r] = f2bf(acc[r]);
    }
}

// =====================================================================
// Kernel 2 (fused): q-chunk (ROWS x 512) -> LDS (bf16) -> scores chunk (ROWS x 128)
// grid(HW/ROWS = 128, B = 16), block(256) = 8 waves, LDS 32 KB
// =====================================================================
__global__ void __launch_bounds__(256)
q_scores_kernel(const float* __restrict__ x, const __bf16* __restrict__ wqt,
                const __bf16* __restrict__ kbf, float* __restrict__ scores)
{
    __shared__ __bf16 qs[ROWS * P_];   // 32 KB

    const int b    = blockIdx.y;
    const int row0 = blockIdx.x * ROWS;
    const int wave = threadIdx.x >> 5;
    const int lane = threadIdx.x & 31;
    const int mr = lane & 15, h = lane >> 4;
    const float* __restrict__ xb = x + (size_t)b * C_ * HW_;

    // ---- Phase 1: q = x^T @ Wq for 32 rows; wave -> (mt = w>>2, 8 P-tiles) ----
    {
        const int mt  = wave >> 2;          // 0..1  (16-row subtile)
        const int pt0 = (wave & 3) * 8;     // 8 column tiles per wave
        const int m   = row0 + mt * 16 + mr;
        v8f acc[8] = {};
        for (int k0 = 0; k0 < C_; k0 += 32) {
            BFrag a;
#pragma unroll
            for (int j = 0; j < 8; ++j) {   // x is (C, HW): gather with stride HW
                a.e[j]     = f2bf(xb[(size_t)(k0 + h * 8 + j) * HW_ + m]);
                a.e[8 + j] = f2bf(xb[(size_t)(k0 + 16 + h * 8 + j) * HW_ + m]);
            }
            __builtin_prefetch(&wqt[(size_t)(pt0 * 16 + mr) * C_ + k0 + 32], 0, 0);
#pragma unroll
            for (int p = 0; p < 8; ++p) {
                BFrag bb;
                const uint4* bp =
                    (const uint4*)(wqt + (size_t)((pt0 + p) * 16 + mr) * C_ + k0 + h * 8);
                bb.u[0] = bp[0];
                bb.u[1] = bp[2];
                acc[p] = wmma_bf16(a, bb, acc[p]);
            }
        }
#pragma unroll
        for (int p = 0; p < 8; ++p)
#pragma unroll
            for (int r = 0; r < 8; ++r)
                qs[(size_t)(mt * 16 + h * 8 + r) * P_ + (pt0 + p) * 16 + mr] = f2bf(acc[p][r]);
    }
    __syncthreads();

    // ---- Phase 2: scores = q @ k^T * scale ; 2x8 tiles, 2 per wave, K = 512 ----
    {
        const __bf16* __restrict__ kb = kbf + (size_t)b * N_ * P_;
        const float scale = 0.044194173824159216f;   // 1/sqrt(512)
        const int mt  = wave >> 2;
        const int nt0 = (wave & 3) * 2;
        v8f acc[2] = {};
        for (int k0 = 0; k0 < P_; k0 += 32) {
            BFrag a;
            a.u[0] = *(const uint4*)&qs[(size_t)(mt * 16 + mr) * P_ + k0 + h * 8];
            a.u[1] = *(const uint4*)&qs[(size_t)(mt * 16 + mr) * P_ + k0 + 16 + h * 8];
#pragma unroll
            for (int i = 0; i < 2; ++i) {
                BFrag bb;
                const uint4* bp =
                    (const uint4*)(kb + (size_t)((nt0 + i) * 16 + mr) * P_ + k0 + h * 8);
                bb.u[0] = bp[0];
                bb.u[1] = bp[2];
                acc[i] = wmma_bf16(a, bb, acc[i]);
            }
        }
#pragma unroll
        for (int i = 0; i < 2; ++i)
#pragma unroll
            for (int r = 0; r < 8; ++r)
                scores[(size_t)b * HW_ * N_ + (size_t)(row0 + mt * 16 + h * 8 + r) * N_
                       + (nt0 + i) * 16 + mr] = acc[i][r] * scale;
    }
}

// =====================================================================
// Kernel 3a: per-(b, chunk, n) partial online-softmax stats over 512 rows
// grid(B, SCH), block(128): thread owns column n, coalesced sweeps
// =====================================================================
__global__ void __launch_bounds__(128)
softmax_partial(const float* __restrict__ scores, float* __restrict__ pm,
                float* __restrict__ ps)
{
    const int b = blockIdx.x, c = blockIdx.y, n = threadIdx.x;
    const float* __restrict__ s =
        scores + (size_t)b * HW_ * N_ + (size_t)c * (HW_ / SCH) * N_ + n;
    float m = -3.0e38f;
    for (int i = 0; i < HW_ / SCH; ++i) m = fmaxf(m, s[(size_t)i * N_]);
    float sum = 0.f;
    for (int i = 0; i < HW_ / SCH; ++i) sum += __expf(s[(size_t)i * N_] - m);
    pm[(b * SCH + c) * N_ + n] = m;
    ps[(b * SCH + c) * N_ + n] = sum;
}

// =====================================================================
// Kernel 3b: combine chunk stats: M = max m_c ; S = sum s_c * exp(m_c - M)
// grid(B), block(128)
// =====================================================================
__global__ void __launch_bounds__(128)
softmax_combine(const float* __restrict__ pm, const float* __restrict__ ps,
                float* __restrict__ mx, float* __restrict__ rs)
{
    const int b = blockIdx.x, n = threadIdx.x;
    float M = -3.0e38f;
#pragma unroll
    for (int c = 0; c < SCH; ++c) M = fmaxf(M, pm[(b * SCH + c) * N_ + n]);
    float S = 0.f;
#pragma unroll
    for (int c = 0; c < SCH; ++c)
        S += ps[(b * SCH + c) * N_ + n] * __expf(pm[(b * SCH + c) * N_ + n] - M);
    mx[b * N_ + n] = M;
    rs[b * N_ + n] = 1.0f / S;
}

// =====================================================================
// Kernel 4 (fused): attn@v chunk (ROWS x 512) -> LDS -> @Wout + residual
// softmax applied on the fly while building the A fragment (exp built once
// per k-step, reused by 8 accumulators).
// grid(HW/ROWS = 128, B = 16), block(256) = 8 waves, LDS 33 KB
// =====================================================================
__global__ void __launch_bounds__(256)
av_out_kernel(const float* __restrict__ scores, const float* __restrict__ mx,
              const float* __restrict__ rs, const __bf16* __restrict__ vt,
              const __bf16* __restrict__ wot, const float* __restrict__ x,
              float* __restrict__ out)
{
    __shared__ __bf16 os[ROWS * P_];   // 32 KB
    __shared__ float smx[N_], srs[N_];

    const int b    = blockIdx.y;
    const int row0 = blockIdx.x * ROWS;
    const int wave = threadIdx.x >> 5;
    const int lane = threadIdx.x & 31;
    const int mr = lane & 15, h = lane >> 4;

    if (threadIdx.x < N_) {
        smx[threadIdx.x] = mx[b * N_ + threadIdx.x];
        srs[threadIdx.x] = rs[b * N_ + threadIdx.x];
    }
    __syncthreads();

    // ---- Phase 1: ob = softmax(scores) @ v ; K = 128 (4 wmma steps) ----
    {
        const int mt  = wave >> 2;
        const int pt0 = (wave & 3) * 8;
        const float* __restrict__ srow =
            scores + (size_t)b * HW_ * N_ + (size_t)(row0 + mt * 16 + mr) * N_;
        const __bf16* __restrict__ vb = vt + (size_t)b * P_ * N_;
        v8f acc[8] = {};
        for (int k0 = 0; k0 < N_; k0 += 32) {
            BFrag a;
#pragma unroll
            for (int j = 0; j < 8; ++j) {
                const int k1 = k0 + h * 8 + j;
                const int k2 = k0 + 16 + h * 8 + j;
                a.e[j]     = f2bf(__expf(srow[k1] - smx[k1]) * srs[k1]);
                a.e[8 + j] = f2bf(__expf(srow[k2] - smx[k2]) * srs[k2]);
            }
#pragma unroll
            for (int p = 0; p < 8; ++p) {
                BFrag bb;
                const uint4* bp =
                    (const uint4*)(vb + (size_t)((pt0 + p) * 16 + mr) * N_ + k0 + h * 8);
                bb.u[0] = bp[0];
                bb.u[1] = bp[2];
                acc[p] = wmma_bf16(a, bb, acc[p]);
            }
        }
#pragma unroll
        for (int p = 0; p < 8; ++p)
#pragma unroll
            for (int r = 0; r < 8; ++r)
                os[(size_t)(mt * 16 + h * 8 + r) * P_ + (pt0 + p) * 16 + mr] = f2bf(acc[p][r]);
    }
    __syncthreads();

    // ---- Phase 2: out = x + ob @ Wout, stored back in (B,C,H,W) layout ----
    {
        const int mt  = wave >> 2;
        const int ct0 = (wave & 3) * 8;
        v8f acc[8] = {};
        for (int k0 = 0; k0 < P_; k0 += 32) {
            BFrag a;
            a.u[0] = *(const uint4*)&os[(size_t)(mt * 16 + mr) * P_ + k0 + h * 8];
            a.u[1] = *(const uint4*)&os[(size_t)(mt * 16 + mr) * P_ + k0 + 16 + h * 8];
            __builtin_prefetch(&wot[(size_t)(ct0 * 16 + mr) * P_ + k0 + 32], 0, 0);
#pragma unroll
            for (int p = 0; p < 8; ++p) {
                BFrag bb;
                const uint4* bp =
                    (const uint4*)(wot + (size_t)((ct0 + p) * 16 + mr) * P_ + k0 + h * 8);
                bb.u[0] = bp[0];
                bb.u[1] = bp[2];
                acc[p] = wmma_bf16(a, bb, acc[p]);
            }
        }
#pragma unroll
        for (int p = 0; p < 8; ++p) {
            const int c = (ct0 + p) * 16 + mr;
#pragma unroll
            for (int r = 0; r < 8; ++r) {
                const int m = row0 + mt * 16 + h * 8 + r;
                const size_t idx = (size_t)b * C_ * HW_ + (size_t)c * HW_ + m;
                out[idx] = x[idx] + acc[p][r];
            }
        }
    }
}

// =====================================================================
// launcher
// =====================================================================
extern "C" void kernel_launch(void* const* d_in, const int* in_sizes, int n_in,
                              void* d_out, int out_size, void* d_ws, size_t ws_size,
                              hipStream_t stream)
{
    (void)in_sizes; (void)n_in; (void)out_size; (void)ws_size;
    const float* x    = (const float*)d_in[0];
    const float* y    = (const float*)d_in[1];
    const float* Wq   = (const float*)d_in[2];
    const float* Wk   = (const float*)d_in[3];
    const float* Wv   = (const float*)d_in[4];
    const float* Wout = (const float*)d_in[5];
    float* out = (float*)d_out;

    char* ws = (char*)d_ws;
    // workspace layout (bytes):
    //   kbf   : [0,         2 MB)     bf16 (B,N,P)
    //   vt    : [2 MB,      4 MB)     bf16 (B,P,N)
    //   mx    : [4194304,   +8 KB)    f32  (B,N)
    //   rs    : [4202496,   +8 KB)    f32  (B,N)
    //   scores: [4210688,   +32 MB)   f32  (B,HW,N)
    //   wqt   : [37765120,  +512 KB)  bf16 (P,C)  n-major
    //   wkt   : [38289408,  +1 MB)    bf16 (P,D)  n-major
    //   wvt   : [39337984,  +1 MB)    bf16 (P,D)  n-major
    //   wot   : [40386560,  +512 KB)  bf16 (C,P)  n-major
    //   pm    : [40910848,  +64 KB)   f32  (B,SCH,N)
    //   ps    : [40976384,  +64 KB)   f32  (B,SCH,N)   total ~41 MB
    __bf16* kbf = (__bf16*)(ws);
    __bf16* vt  = (__bf16*)(ws + 2097152);
    float*  mx  = (float*)(ws + 4194304);
    float*  rs  = (float*)(ws + 4202496);
    float*  sc  = (float*)(ws + 4210688);
    __bf16* wqt = (__bf16*)(ws + 37765120);
    __bf16* wkt = (__bf16*)(ws + 38289408);
    __bf16* wvt = (__bf16*)(ws + 39337984);
    __bf16* wot = (__bf16*)(ws + 40386560);
    float*  pm  = (float*)(ws + 40910848);
    float*  ps  = (float*)(ws + 40976384);

    // weight transpose+convert (tiny, tiled/coalesced)
    transpose_cvt<<<dim3(C_ / 32, P_ / 32), 256, 0, stream>>>(Wq,   wqt, C_, P_);
    transpose_cvt<<<dim3(D_ / 32, P_ / 32), 256, 0, stream>>>(Wk,   wkt, D_, P_);
    transpose_cvt<<<dim3(D_ / 32, P_ / 32), 256, 0, stream>>>(Wv,   wvt, D_, P_);
    transpose_cvt<<<dim3(P_ / 32, C_ / 32), 256, 0, stream>>>(Wout, wot, P_, C_);

    kv_kernel<<<dim3(P_ / 16, N_ / 16, B_ * 2), 32, 0, stream>>>(y, wkt, wvt, kbf, vt);
    q_scores_kernel<<<dim3(HW_ / ROWS, B_), 256, 0, stream>>>(x, wqt, kbf, sc);
    softmax_partial<<<dim3(B_, SCH), N_, 0, stream>>>(sc, pm, ps);
    softmax_combine<<<B_, N_, 0, stream>>>(pm, ps, mx, rs);
    av_out_kernel<<<dim3(HW_ / ROWS, B_), 256, 0, stream>>>(sc, mx, rs, vt, wot, x, out);
}
